// Amiya_63230508532028
// MI455X (gfx1250) — matrix operand order repaired and checked by reading, not compiled, not run on hardware
//
#include <hip/hip_runtime.h>
#include <hip/hip_bf16.h>

// ---------------------------------------------------------------------------
// Config
// ---------------------------------------------------------------------------
#define AMK_ASYNC 1        // global_load_async_to_lds_b128 for full B tiles
#if __has_builtin(__builtin_amdgcn_tensor_load_to_lds)
#define AMK_TDM 1          // tensor_load_to_lds (TDM) for full A tiles
#else
#define AMK_TDM 0
#endif

typedef unsigned short u16;
typedef __bf16 v16bf __attribute__((ext_vector_type(16)));
typedef float  v8f   __attribute__((ext_vector_type(8)));
typedef unsigned v4u __attribute__((ext_vector_type(4)));
typedef int      v4i __attribute__((ext_vector_type(4)));
typedef int      v8i __attribute__((ext_vector_type(8)));

#define NG    2000
#define NR    3000
#define DE    512
#define FM    13
#define NH    4
#define BSZ   16
#define NSPOT 8192
#define NMET  (BSZ * NR)

__device__ __forceinline__ float gelu_f(float x) {
    return 0.5f * x * (1.0f + erff(x * 0.70710678118654752f));
}

__device__ __forceinline__ u16 f2bf(float f) {
    unsigned u = __float_as_uint(f);
    u += 0x7FFFu + ((u >> 16) & 1u);   // round-to-nearest-even
    return (u16)(u >> 16);
}

// f32 -> bf16 conversion, float4 -> packed uint2 (all uses are 4-divisible, 16B aligned)
__global__ void amk_cvt_bf16(const float* __restrict__ in, u16* __restrict__ out, size_t n)
{
    size_t i4 = ((size_t)blockIdx.x * blockDim.x + threadIdx.x) * 4;
    if (i4 + 4 <= n) {
        float4 v = *(const float4*)(in + i4);
        uint2 p;
        p.x = (unsigned)f2bf(v.x) | ((unsigned)f2bf(v.y) << 16);
        p.y = (unsigned)f2bf(v.z) | ((unsigned)f2bf(v.w) << 16);
        *(uint2*)(out + i4) = p;
    } else {
        for (size_t i = i4; i < n; ++i) out[i] = f2bf(in[i]);
    }
}

// ---------------------------------------------------------------------------
// WMMA GEMM:  Y[M,N] = act( X[M,K] @ W[N,K]^T + bias ),  X/W bf16, acc f32.
// mode: 0 = plain, 1 = gelu, 2 = relu ; outBf16: write u16 instead of f32.
// 64x64 tile, 4 waves, BK=32, double-buffered LDS:
//   - full A tiles staged by the Tensor Data Mover (2D D#, LDS pad to LDT=40)
//   - full B tiles staged by global_load_async_to_lds_b128
//   - partial tiles by vectorized (uint4) sync loads (every K here is 8-divisible)
// Loads for tile k+1 are issued before computing tile k; waits (TENSORcnt /
// ASYNCcnt) land after the WMMAs so the DMA latency hides under compute.
// ---------------------------------------------------------------------------
#define TKK 32
#define LDT 40   // padded LDS row stride (bf16 elems); 80B rows, 16B aligned

__global__ void __launch_bounds__(128)
amk_gemm_wmma(const u16* __restrict__ X, const u16* __restrict__ W,
              const float* __restrict__ bias, void* __restrict__ Yv,
              int M, int N, int K, int mode, int outBf16)
{
    __shared__ u16 As[2][64 * LDT];
    __shared__ u16 Bs[2][64 * LDT];
    const int tid   = threadIdx.x;
    const int lane  = tid & 31;
    const int wv    = tid >> 5;
    const int tileN = blockIdx.x * 64;
    const int tileM = blockIdx.y * 64;

    v8f acc[4];
#pragma unroll
    for (int i = 0; i < 4; ++i)
#pragma unroll
        for (int j = 0; j < 8; ++j) acc[i][j] = 0.0f;

    const int numK = (K + TKK - 1) / TKK;

    // flags: 1 = TDM issued (TENSORcnt), 2 = async issued (ASYNCcnt)
    auto stageA = [&](int kt, int buf) -> int {
        const int k0 = kt * TKK;
        const bool full = (k0 + TKK <= K) && (tileM + 64 <= M);
#if AMK_TDM
        if (full) {
            if (wv == 0) {
                unsigned lds = (unsigned)(size_t)&As[buf][0];
                unsigned long long ga =
                    (unsigned long long)(size_t)(X + (size_t)tileM * K + k0);
                v4u g0;
                g0[0] = 1u;                                   // count=1 (user D#)
                g0[1] = lds;                                  // lds_addr
                g0[2] = (unsigned)(ga & 0xFFFFFFFFu);         // global_addr[31:0]
                g0[3] = (unsigned)((ga >> 32) & 0x01FFFFFFu)  // global_addr[56:32]
                        | (2u << 30);                         // type=2 (image)
                v8i g1;
                g1[0] = (int)((1u << 16)      // data_size = 2 bytes
                            | (1u << 20)      // pad_enable
                            | (3u << 22)      // pad_interval: 16 DWORDs (64B row)
                            | (3u << 25));    // pad_amount: 4 DWORDs (16B -> LDT=40)
                g1[1] = (int)(((unsigned)K & 0xFFFFu) << 16);             // tensor_dim0 lo
                g1[2] = (int)((((unsigned)K >> 16) & 0xFFFFu) | (64u << 16)); // dim0 hi | tensor_dim1 lo
                g1[3] = (int)(32u << 16);     // tensor_dim1 hi=0 | tile_dim0=32
                g1[4] = (int)64u;             // tile_dim1=64 | tile_dim2=0
                g1[5] = (int)(unsigned)K;     // tensor_dim0_stride lo (elems)
                g1[6] = 0;                    // stride hi | dim1_stride lo
                g1[7] = 0;
                v4i z4 = {0, 0, 0, 0};
#if defined(__clang_major__) && (__clang_major__ >= 23)
                v8i z8 = {0, 0, 0, 0, 0, 0, 0, 0};
                __builtin_amdgcn_tensor_load_to_lds(g0, g1, z4, z4, z8, 0);
#else
                __builtin_amdgcn_tensor_load_to_lds(g0, g1, z4, z4, 0);
#endif
            }
            return 1;
        }
#endif
        for (int c = tid; c < 256; c += 128) {
            const int r = c >> 2, q = c & 3;
            const int gm = tileM + r;
            uint4 val = {0u, 0u, 0u, 0u};
            if (gm < M && (k0 + q * 8 + 8) <= K)
                val = *(const uint4*)(X + (size_t)gm * K + k0 + q * 8);
            *(uint4*)(&As[buf][r * LDT + q * 8]) = val;
        }
        return 0;
    };

    auto stageB = [&](int kt, int buf) -> int {
        const int k0 = kt * TKK;
        const bool full = (k0 + TKK <= K) && (tileN + 64 <= N);
#if AMK_ASYNC
        if (full) {
            for (int c = tid; c < 256; c += 128) {           // 256 x 16B chunks
                const int r = c >> 2, q = c & 3;
                const u16* gp = W + (size_t)(tileN + r) * K + k0 + q * 8;
                unsigned lds = (unsigned)(size_t)&Bs[buf][r * LDT + q * 8];
                unsigned long long ga = (unsigned long long)(size_t)gp;
                asm volatile("global_load_async_to_lds_b128 %0, %1, off"
                             :: "v"(lds), "v"(ga) : "memory");
            }
            return 2;
        }
#endif
        for (int c = tid; c < 256; c += 128) {
            const int r = c >> 2, q = c & 3;
            const int gn = tileN + r;
            uint4 val = {0u, 0u, 0u, 0u};
            if (gn < N && (k0 + q * 8 + 8) <= K)
                val = *(const uint4*)(W + (size_t)gn * K + k0 + q * 8);
            *(uint4*)(&Bs[buf][r * LDT + q * 8]) = val;
        }
        return 0;
    };

    auto waitStage = [&](int flags) {
#if AMK_ASYNC
        if (flags & 2) asm volatile("s_wait_asynccnt 0x0" ::: "memory");
#endif
#if AMK_TDM
        if (flags & 1) __builtin_amdgcn_s_wait_tensorcnt((short)0);
#endif
        (void)flags;
    };

    // prologue: stage tile 0
    int fl = stageA(0, 0) | stageB(0, 0);
    waitStage(fl);
    __syncthreads();

    for (int kt = 0; kt < numK; ++kt) {
        const int cur = kt & 1;
        int nfl = 0;
        if (kt + 1 < numK)
            nfl = stageA(kt + 1, cur ^ 1) | stageB(kt + 1, cur ^ 1);
        if (kt + 2 < numK) {   // speculative prefetch two tiles ahead
            int pr = tileM + (tid >> 1); if (pr >= M) pr = M - 1;
            __builtin_prefetch(&X[(size_t)pr * K + (kt + 2) * TKK], 0, 1);
        }

        // A fragment: lane holds row M=lane%16, K pair-packed per ISA layout.
        const u16* Ab = &As[cur][0];
        const u16* Bbs = &Bs[cur][0];
        union { v16bf v; unsigned u[8]; } afr;
        {
            const int m = 16 * wv + (lane & 15);
            const int h = lane >> 4;
            const u16* rp = Ab + m * LDT;
#pragma unroll
            for (int j = 0; j < 8; ++j) {
                const int kb = ((j < 4) ? 2 * j : 2 * j + 8) + 8 * h;
                afr.u[j] = *(const unsigned*)(rp + kb);
            }
        }
#pragma unroll
        for (int nt = 0; nt < 4; ++nt) {
            // B fragment: lane holds col N=lane%16, K pair-packed per ISA layout (same as A).
            union { v16bf v; unsigned u[8]; } bfr;
            const int n = 16 * nt + (lane & 15);
            const u16* rp = Bbs + n * LDT + 8 * (lane >> 4);
#pragma unroll
            for (int j = 0; j < 8; ++j) bfr.u[j] = *(const unsigned*)(rp + ((j < 4) ? 2 * j : 2 * j + 8));
            acc[nt] = __builtin_amdgcn_wmma_f32_16x16x32_bf16(
                false, afr.v, false, bfr.v, (short)0, acc[nt], false, false);
        }
        waitStage(nfl);        // next tile's DMA completes under this tile's WMMAs
        __syncthreads();
    }

    // Epilogue. C/D layout: lanes 0-15 rows 0..7, lanes 16-31 rows 8..15.
    const int mrow0 = tileM + 16 * wv + 8 * (lane >> 4);
#pragma unroll
    for (int nt = 0; nt < 4; ++nt) {
        const int gn = tileN + 16 * nt + (lane & 15);
        if (gn >= N) continue;
        const float bv = bias ? bias[gn] : 0.0f;
#pragma unroll
        for (int r = 0; r < 8; ++r) {
            const int gm = mrow0 + r;
            if (gm >= M) continue;
            float v = acc[nt][r] + bv;
            if (mode == 1)      v = gelu_f(v);
            else if (mode == 2) v = fmaxf(v, 0.0f);
            if (outBf16) ((u16*)Yv)[(size_t)gm * N + gn] = f2bf(v);
            else         ((float*)Yv)[(size_t)gm * N + gn] = v;
        }
    }
}

// ---------------------------------------------------------------------------
// GraphNorm (batched, contiguous segments): stats then apply (bf16 out).
// var = E[x^2] + (a^2 - 2a) * m^2   (out = x - a*m, var = mean(out^2))
// ---------------------------------------------------------------------------
__global__ void amk_seg_stats(const float* __restrict__ X, const float* __restrict__ a,
                              float* __restrict__ mean, float* __restrict__ rstd,
                              int segRows, int C)
{
    const int c = blockIdx.x * 256 + threadIdx.x;
    const int seg = blockIdx.y;
    if (c >= C) return;
    const float* p = X + (size_t)seg * segRows * C + c;
    float s1 = 0.f, s2 = 0.f;
    for (int r = 0; r < segRows; ++r) { float v = p[(size_t)r * C]; s1 += v; s2 += v * v; }
    const float inv = 1.0f / (float)segRows;
    const float m = s1 * inv, ex2 = s2 * inv, av = a[c];
    const float var = ex2 + (av * av - 2.0f * av) * m * m;
    mean[seg * C + c] = m;
    rstd[seg * C + c] = rsqrtf(var + 1e-5f);
}

__global__ void amk_seg_apply(const float* __restrict__ X, u16* __restrict__ Y,
                              const float* __restrict__ a, const float* __restrict__ w,
                              const float* __restrict__ b, const float* __restrict__ mean,
                              const float* __restrict__ rstd, int segRows, int C, size_t n)
{
    size_t idx = (size_t)blockIdx.x * blockDim.x + threadIdx.x;
    if (idx >= n) return;
    const int c = (int)(idx % C);
    const int seg = (int)((idx / C) / segRows);
    float v = X[idx];
    v = w[c] * (v - a[c] * mean[seg * C + c]) * rstd[seg * C + c] + b[c];
    Y[idx] = f2bf(v);
}

// GraphNorm1 (single graph): column partial sums -> finalize -> apply+gelu(bf16)
__global__ void amk_col_partial(const float* __restrict__ X, int M, int C, int rowsPer,
                                float* __restrict__ S1, float* __restrict__ S2)
{
    const int c = blockIdx.x * 256 + threadIdx.x;
    if (c >= C) return;
    int r0 = blockIdx.y * rowsPer;
    int r1 = r0 + rowsPer; if (r1 > M) r1 = M;
    float s1 = 0.f, s2 = 0.f;
    for (int r = r0; r < r1; ++r) { float v = X[(size_t)r * C + c]; s1 += v; s2 += v * v; }
    atomicAdd(&S1[c], s1);
    atomicAdd(&S2[c], s2);
}

__global__ void amk_col_finalize(const float* __restrict__ S1, const float* __restrict__ S2,
                                 const float* __restrict__ a, float* __restrict__ mean,
                                 float* __restrict__ rstd, int C, float invM)
{
    const int c = blockIdx.x * 256 + threadIdx.x;
    if (c >= C) return;
    const float m = S1[c] * invM, ex2 = S2[c] * invM, av = a[c];
    const float var = ex2 + (av * av - 2.0f * av) * m * m;
    mean[c] = m;
    rstd[c] = rsqrtf(var + 1e-5f);
}

__global__ void amk_gn1_apply(const float* __restrict__ X, u16* __restrict__ Y,
                              const float* __restrict__ a, const float* __restrict__ w,
                              const float* __restrict__ b, const float* __restrict__ mean,
                              const float* __restrict__ rstd, size_t n, int C)
{
    size_t idx = (size_t)blockIdx.x * blockDim.x + threadIdx.x;
    if (idx >= n) return;
    const int c = (int)(idx % C);
    float v = X[idx];
    v = w[c] * (v - a[c] * mean[c]) * rstd[c] + b[c];
    Y[idx] = f2bf(gelu_f(v));
}

// ---------------------------------------------------------------------------
// Graph / elementwise helpers (f32, atomic-bound)
// ---------------------------------------------------------------------------
__global__ void amk_scatter_add(const float* __restrict__ X, float* __restrict__ Out,
                                const int* __restrict__ src, const int* __restrict__ dst,
                                int E, int F)
{
    const int f = blockIdx.x * 256 + threadIdx.x;
    const int e = blockIdx.y;
    if (f >= F || e >= E) return;
    atomicAdd(&Out[(size_t)dst[e] * F + f], X[(size_t)src[e] * F + f]);
}

__global__ void amk_eadd(const float* __restrict__ A, const float* __restrict__ B,
                         float* __restrict__ O, size_t n)
{
    size_t idx = (size_t)blockIdx.x * blockDim.x + threadIdx.x;
    if (idx >= n) return;
    O[idx] = A[idx] + B[idx];
}

__global__ void amk_seg_reduce(const float* __restrict__ X, float* __restrict__ O,
                               int segRows, int C, int accum)
{
    const int c = blockIdx.x * 256 + threadIdx.x;
    const int seg = blockIdx.y;
    if (c >= C) return;
    const float* p = X + (size_t)seg * segRows * C + c;
    float s = 0.f;
    for (int r = 0; r < segRows; ++r) s += p[(size_t)r * C];
    const float v = gelu_f(s);
    if (accum) O[seg * C + c] += v; else O[seg * C + c] = v;
}

__global__ void amk_build_blockT(const float* __restrict__ Xrxn, float* __restrict__ BlkT)
{
    const int idx = blockIdx.x * 256 + threadIdx.x;   // NR*FM
    if (idx >= NR * FM) return;
    const int r = idx / FM, c = idx % FM;
    BlkT[idx] = Xrxn[(size_t)c * NR + r];             // x_rxn[:FM,:].T
}

__global__ void amk_tile_xmet(const float* __restrict__ BlkT, float* __restrict__ Xm, int total)
{
    const int idx = blockIdx.x * 256 + threadIdx.x;
    if (idx >= total) return;
    const int node = idx / FM, c = idx % FM;
    Xm[idx] = BlkT[(node % NR) * FM + c];
}

// ---------------------------------------------------------------------------
// SuperGAT (MX attention, heads=4, F=13)
// ---------------------------------------------------------------------------
__global__ void amk_sgat_h(const float* __restrict__ In, const float* __restrict__ W,
                           float* __restrict__ H, int Nnodes)
{
    const int idx = blockIdx.x * 256 + threadIdx.x;   // Nnodes * NH*FM
    if (idx >= Nnodes * NH * FM) return;
    const int node = idx / (NH * FM), o = idx % (NH * FM);
    const float* xr = In + (size_t)node * FM;
    const float* wr = W + (size_t)o * FM;
    float s = 0.f;
#pragma unroll
    for (int c = 0; c < FM; ++c) s += xr[c] * wr[c];
    H[idx] = s;
}

__device__ __forceinline__ unsigned ford_enc(float f) {
    unsigned u = __float_as_uint(f);
    return (u & 0x80000000u) ? ~u : (u | 0x80000000u);
}
__device__ __forceinline__ float ford_dec(unsigned u) {
    return (u & 0x80000000u) ? __uint_as_float(u & 0x7FFFFFFFu) : __uint_as_float(~u);
}

__global__ void amk_edge_alpha(const float* __restrict__ H, const int* __restrict__ src,
                               const int* __restrict__ dst, const float* __restrict__ attl,
                               const float* __restrict__ attr, float* __restrict__ alpha,
                               unsigned* __restrict__ amax, int E)
{
    const int idx = blockIdx.x * 256 + threadIdx.x;
    if (idx >= E * NH) return;
    const int e = idx >> 2, hd = idx & 3;
    const int s = src[e], d = dst[e];
    const float* xj = H + (size_t)s * (NH * FM) + hd * FM;
    const float* xi = H + (size_t)d * (NH * FM) + hd * FM;
    float lg = 0.f, al = 0.f;
#pragma unroll
    for (int c = 0; c < FM; ++c) {
        lg += xi[c] * xj[c];
        al += xj[c] * attl[hd * FM + c] + xi[c] * attr[hd * FM + c];
    }
    al = al * (1.0f / (1.0f + expf(-lg)));
    al = (al >= 0.f) ? al : 0.2f * al;      // leaky_relu 0.2
    alpha[idx] = al;
    atomicMax(&amax[(size_t)d * NH + hd], ford_enc(al));
}

__global__ void amk_edge_exp(const int* __restrict__ dst, float* __restrict__ alpha,
                             const unsigned* __restrict__ amax, float* __restrict__ denom, int E)
{
    const int idx = blockIdx.x * 256 + threadIdx.x;
    if (idx >= E * NH) return;
    const int e = idx >> 2, hd = idx & 3;
    const int d = dst[e];
    const float ex = expf(alpha[idx] - ford_dec(amax[(size_t)d * NH + hd]));
    alpha[idx] = ex;
    atomicAdd(&denom[(size_t)d * NH + hd], ex);
}

__global__ void amk_edge_out(const float* __restrict__ H, const int* __restrict__ src,
                             const int* __restrict__ dst, const float* __restrict__ alpha,
                             const float* __restrict__ denom, float* __restrict__ Out, int E)
{
    const int idx = blockIdx.x * 256 + threadIdx.x;
    if (idx >= E * NH) return;
    const int e = idx >> 2, hd = idx & 3;
    const int s = src[e], d = dst[e];
    const float w = alpha[idx] / (denom[(size_t)d * NH + hd] + 1e-16f);
    const float* xj = H + (size_t)s * (NH * FM) + hd * FM;
#pragma unroll
    for (int c = 0; c < FM; ++c)
        atomicAdd(&Out[(size_t)d * (NH * FM) + hd * FM + c], xj[c] * w);
}

__global__ void amk_node_me(const float* __restrict__ Agg, const float* __restrict__ bias,
                            const float* __restrict__ AddX, float* __restrict__ Out, int Nnodes)
{
    const int idx = blockIdx.x * 256 + threadIdx.x;
    if (idx >= Nnodes * FM) return;
    const int node = idx / FM, c = idx % FM;
    float s = 0.f;
#pragma unroll
    for (int h = 0; h < NH; ++h) s += Agg[(size_t)node * (NH * FM) + h * FM + c];
    float v = gelu_f(0.25f * s + bias[c]);
    if (AddX) v += AddX[idx];
    Out[idx] = v;
}

__global__ void amk_smax_aggr(const float* __restrict__ X, const float* __restrict__ tptr,
                              float* __restrict__ Out, int Nnodes, int accum)
{
    const int node = blockIdx.x * 256 + threadIdx.x;
    if (node >= Nnodes) return;
    const float t = tptr[0];
    const float* xr = X + (size_t)node * FM;
    float mx = -3.0e38f;
#pragma unroll
    for (int c = 0; c < FM; ++c) mx = fmaxf(mx, t * xr[c]);
    float ds = 0.f, num = 0.f;
#pragma unroll
    for (int c = 0; c < FM; ++c) {
        const float e = expf(t * xr[c] - mx);
        ds += e; num += e * xr[c];
    }
    const float v = gelu_f(num / ds);
    if (accum) Out[node] += v; else Out[node] = v;
}

__global__ void amk_concat2(const float* __restrict__ A, const float* __restrict__ B,
                            u16* __restrict__ O, int M, int C)
{
    const int idx = blockIdx.x * 256 + threadIdx.x;
    if (idx >= M * 2 * C) return;
    const int row = idx / (2 * C), col = idx % (2 * C);
    O[idx] = f2bf((col < C) ? A[row * C + col] : B[row * C + col - C]);
}

// ---------------------------------------------------------------------------
// Host orchestration
// ---------------------------------------------------------------------------
extern "C" void kernel_launch(void* const* d_in, const int* in_sizes, int n_in,
                              void* d_out, int out_size, void* d_ws, size_t ws_size,
                              hipStream_t stream)
{
    (void)n_in; (void)out_size; (void)ws_size;
    const float* X   = (const float*)d_in[0];
    const float* P   = (const float*)d_in[1];
    const int*   ei  = (const int*)d_in[3];
    const int*   mei = (const int*)d_in[5];
    const int EST = in_sizes[3] / 2;
    const int EM  = in_sizes[5] / 2;
    const int* st_src = ei;
    const int* st_dst = ei + EST;
    const int* m_src  = mei;
    const int* m_dst  = mei + EM;

    // ---- param walk: jax pytree flattening = dict keys sorted alphabetically ----
    size_t off = 0;
    auto take = [&](size_t n) { const float* r = P + off; off += n; return r; };
    struct Lin { const float* b; const float* w; };
    struct Nrm { const float* a; const float* b; const float* w; };
    auto takeLin = [&](int in, int out) { Lin l; l.b = take(out); l.w = take((size_t)out * in); return l; };
    auto takeNrm = [&](int d) { Nrm n; n.a = take(d); n.b = take(d); n.w = take(d); return n; };

    Lin e0 = takeLin(1024, 128), e1 = takeLin(128, 512), e2 = takeLin(512, 1024), e3 = takeLin(1024, 3000);
    Nrm en0 = takeNrm(128), en1 = takeNrm(512), en2 = takeNrm(1024);
    Lin grm = takeLin(NG, NR);
    Lin g00 = takeLin(3000, 1024), g01 = takeLin(1024, 512), g02 = takeLin(512, 128), g03 = takeLin(128, 512);
    Nrm gn00 = takeNrm(1024), gn01 = takeNrm(512), gn02 = takeNrm(128);
    Lin g10 = takeLin(512, 256), g11 = takeLin(256, 128), g12 = takeLin(128, 256), g13 = takeLin(256, 512);
    Nrm gn10 = takeNrm(256), gn11 = takeNrm(128), gn12 = takeNrm(256);
    Lin idl = takeLin(NG, DE);
    (void)takeLin(NG, NR);            // identity_met (dead in reference)
    Nrm ign = takeNrm(NG);
    Lin mc0 = takeLin(3000, 512), mc1 = takeLin(512, 256), mc2 = takeLin(256, 512);
    Nrm mcn0 = takeNrm(512), mcn1 = takeNrm(256);
    Lin rgm = takeLin(NR, NG);
    const float* attl0 = take(NH * FM); const float* attr0 = take(NH * FM);
    const float* sb0 = take(FM);        const float* sw0 = take(NH * FM * FM);
    const float* attl1 = take(NH * FM); const float* attr1 = take(NH * FM);
    const float* sb1 = take(FM);        const float* sw1 = take(NH * FM * FM);
    const float* t0 = take(1); const float* t1 = take(1);

    // ---- workspace carve-out (units of float; bf16 buffers take n/2 slots) ----
    size_t wo = 0;
    auto walloc = [&](size_t nfloats) { float* p = (float*)d_ws + wo; wo += (nfloats + 63) & ~(size_t)63; return p; };
    float*    A     = walloc((size_t)NSPOT * NR);
    float*    Bb    = walloc((size_t)NSPOT * NR);
    float*    ID    = walloc((size_t)NSPOT * DE);
    float*    Cc    = walloc((size_t)NSPOT * DE);
    float*    STCAT = walloc((size_t)BSZ * DE);
    float*    BLK   = walloc((size_t)NR * FM);
    float*    XMET  = walloc((size_t)NMET * FM);
    float*    H     = walloc((size_t)NMET * NH * FM);
    float*    ALPHA = walloc((size_t)EM * NH);
    unsigned* AMAX  = (unsigned*)walloc((size_t)NMET * NH);
    float*    DEN   = walloc((size_t)NMET * NH);
    float*    AOUT  = walloc((size_t)NMET * NH * FM);
    float*    SMET  = walloc((size_t)NMET * FM);
    float*    ME2   = walloc((size_t)NMET * FM);
    float*    MCAT  = walloc((size_t)BSZ * NR);
    float*    M1    = walloc((size_t)BSZ * NR);
    float*    M2    = walloc((size_t)BSZ * NR);
    float*    S1    = walloc(3072);
    float*    S2    = walloc(3072);
    float*    MEAN  = walloc(3072);
    float*    RSTD  = walloc(3072);
    float*    SGM   = walloc((size_t)BSZ * NG);
    float*    SGR   = walloc((size_t)BSZ * NG);
    // bf16 buffers
    u16* XNh   = (u16*)walloc(((size_t)NSPOT * NG + 1) / 2);
    u16* Wh    = (u16*)walloc(((size_t)NR * NG + 1) / 2);      // weight scratch (max 3000x2000)
    u16* ACTh  = (u16*)walloc(((size_t)NSPOT * NR + 1) / 2);
    u16* ACT2h = (u16*)walloc(((size_t)NSPOT * 1024 + 1) / 2);
    u16* MCATh = (u16*)walloc(((size_t)BSZ * NR + 1) / 2);
    u16* M1h   = (u16*)walloc(((size_t)BSZ * NR + 1) / 2);
    u16* M2h   = (u16*)walloc(((size_t)BSZ * NR + 1) / 2);
    u16* CATh  = (u16*)walloc(((size_t)BSZ * 2 * DE + 1) / 2);
    u16* OUTh  = (u16*)walloc(((size_t)BSZ * NR + 1) / 2);

    const int segRows = NSPOT / BSZ;   // 512

    auto cvt = [&](const float* in, u16* out, size_t n) {
        amk_cvt_bf16<<<dim3((unsigned)((n / 4 + 255) / 256)), 256, 0, stream>>>(in, out, n);
    };
    // convert weights into scratch, then GEMM (bf16 x bf16 -> f32/bf16)
    auto gemm = [&](const u16* Xh, const Lin& L, int Kc, int Nc, void* Yp, int Mr, int mode, int obf) {
        cvt(L.w, Wh, (size_t)Nc * Kc);
        dim3 g((Nc + 63) / 64, (Mr + 63) / 64);
        amk_gemm_wmma<<<g, 128, 0, stream>>>(Xh, Wh, L.b, Yp, Mr, Nc, Kc, mode, obf);
    };
    auto norm_gelu = [&](const float* Xf, u16* Yh, int Mr, int C, const Nrm& N) {
        hipMemsetAsync(S1, 0, C * sizeof(float), stream);
        hipMemsetAsync(S2, 0, C * sizeof(float), stream);
        const int chunks = (Mr >= 1024) ? 32 : 1;
        const int rowsPer = (Mr + chunks - 1) / chunks;
        amk_col_partial<<<dim3((C + 255) / 256, chunks), 256, 0, stream>>>(Xf, Mr, C, rowsPer, S1, S2);
        amk_col_finalize<<<dim3((C + 255) / 256), 256, 0, stream>>>(S1, S2, N.a, MEAN, RSTD, C, 1.0f / (float)Mr);
        const size_t n = (size_t)Mr * C;
        amk_gn1_apply<<<dim3((unsigned)((n + 255) / 256)), 256, 0, stream>>>(Xf, Yh, N.a, N.w, N.b, MEAN, RSTD, n, C);
    };

    // ---- 1) input GraphNorm over batch segments (bf16 out) ----
    amk_seg_stats<<<dim3((NG + 255) / 256, BSZ), 256, 0, stream>>>(X, ign.a, SGM, SGR, segRows, NG);
    {
        const size_t n = (size_t)NSPOT * NG;
        amk_seg_apply<<<dim3((unsigned)((n + 255) / 256)), 256, 0, stream>>>(
            X, XNh, ign.a, ign.w, ign.b, SGM, SGR, segRows, NG, n);
    }

    // ---- 2) x_id = gelu(xn @ identity^T), x_rxn = gelu(xn @ gene_rxn^T) ----
    gemm(XNh, idl, NG, DE, ID, NSPOT, 1, 0);
    gemm(XNh, grm, NG, NR, A, NSPOT, 1, 0);                     // x_rxn (f32) in A

    // met-branch base tile: block = x_rxn[:13,:].T -> (3000,13), tiled x16
    amk_build_blockT<<<dim3((NR * FM + 255) / 256), 256, 0, stream>>>(A, BLK);
    amk_tile_xmet<<<dim3((NMET * FM + 255) / 256), 256, 0, stream>>>(BLK, XMET, NMET * FM);

    // ---- 3) GIN0: gin_in = x_rxn + scatter_sum(x_rxn[src] -> dst) ----
    hipMemcpyAsync(Bb, A, (size_t)NSPOT * NR * sizeof(float), hipMemcpyDeviceToDevice, stream);
    amk_scatter_add<<<dim3((NR + 255) / 256, EST), 256, 0, stream>>>(A, Bb, st_src, st_dst, EST, NR);
    cvt(Bb, ACTh, (size_t)NSPOT * NR);
    gemm(ACTh, g00, 3000, 1024, A, NSPOT, 0, 0);  norm_gelu(A, ACT2h, NSPOT, 1024, gn00);
    gemm(ACT2h, g01, 1024, 512, Bb, NSPOT, 0, 0); norm_gelu(Bb, ACTh, NSPOT, 512, gn01);
    gemm(ACTh, g02, 512, 128, A, NSPOT, 0, 0);    norm_gelu(A, ACT2h, NSPOT, 128, gn02);
    gemm(ACT2h, g03, 128, 512, Bb, NSPOT, 1, 0);                // st = gelu(gin0) (f32)

    // s = st + x_id ; st_cat = gelu(segment_sum(s))
    amk_eadd<<<dim3((unsigned)(((size_t)NSPOT * DE + 255) / 256)), 256, 0, stream>>>(
        Bb, ID, Cc, (size_t)NSPOT * DE);
    amk_seg_reduce<<<dim3((DE + 255) / 256, BSZ), 256, 0, stream>>>(Cc, STCAT, segRows, DE, 0);

    // ---- 4) GIN1 on s ----
    hipMemcpyAsync(A, Cc, (size_t)NSPOT * DE * sizeof(float), hipMemcpyDeviceToDevice, stream);
    amk_scatter_add<<<dim3((DE + 255) / 256, EST), 256, 0, stream>>>(Cc, A, st_src, st_dst, EST, DE);
    cvt(A, ACTh, (size_t)NSPOT * DE);
    gemm(ACTh, g10, 512, 256, Bb, NSPOT, 0, 0);  norm_gelu(Bb, ACT2h, NSPOT, 256, gn10);
    gemm(ACT2h, g11, 256, 128, A, NSPOT, 0, 0);  norm_gelu(A, ACTh, NSPOT, 128, gn11);
    gemm(ACTh, g12, 128, 256, Bb, NSPOT, 0, 0);  norm_gelu(Bb, ACT2h, NSPOT, 256, gn12);
    gemm(ACT2h, g13, 256, 512, A, NSPOT, 1, 0);                 // st2 = gelu(gin1) (f32)
    amk_seg_reduce<<<dim3((DE + 255) / 256, BSZ), 256, 0, stream>>>(A, STCAT, segRows, DE, 1);

    // ---- 5) Met branch: SuperGAT x2 + softmax-aggr ----
    const dim3 gEH((EM * NH + 255) / 256);
    auto supergat = [&](const float* feat, const float* W, const float* attl,
                        const float* attr, const float* bias, const float* addx, float* outme) {
        amk_sgat_h<<<dim3((NMET * NH * FM + 255) / 256), 256, 0, stream>>>(feat, W, H, NMET);
        hipMemsetAsync(AOUT, 0, (size_t)NMET * NH * FM * sizeof(float), stream);
        hipMemsetAsync(AMAX, 0, (size_t)NMET * NH * sizeof(unsigned), stream);
        hipMemsetAsync(DEN, 0, (size_t)NMET * NH * sizeof(float), stream);
        amk_edge_alpha<<<gEH, 256, 0, stream>>>(H, m_src, m_dst, attl, attr, ALPHA, AMAX, EM);
        amk_edge_exp<<<gEH, 256, 0, stream>>>(m_dst, ALPHA, AMAX, DEN, EM);
        amk_edge_out<<<gEH, 256, 0, stream>>>(H, m_src, m_dst, ALPHA, DEN, AOUT, EM);
        amk_node_me<<<dim3((NMET * FM + 255) / 256), 256, 0, stream>>>(AOUT, bias, addx, outme, NMET);
    };
    supergat(XMET, sw0, attl0, attr0, sb0, XMET, SMET);         // s_met = gelu(sgat0)+x_met
    amk_smax_aggr<<<dim3((NMET + 255) / 256), 256, 0, stream>>>(SMET, t0, MCAT, NMET, 0);
    supergat(SMET, sw1, attl1, attr1, sb1, nullptr, ME2);       // me2 = gelu(sgat1)
    amk_smax_aggr<<<dim3((NMET + 255) / 256), 256, 0, stream>>>(ME2, t1, MCAT, NMET, 1);

    // ---- 6) met_compress MLP on (16, 3000) ----
    cvt(MCAT, MCATh, (size_t)BSZ * NR);
    gemm(MCATh, mc0, 3000, 512, M1, BSZ, 0, 0); norm_gelu(M1, M1h, BSZ, 512, mcn0);
    gemm(M1h, mc1, 512, 256, M2, BSZ, 0, 0);    norm_gelu(M2, M2h, BSZ, 256, mcn1);
    gemm(M2h, mc2, 256, 512, M1, BSZ, 1, 0);                    // met_cat_c (f32)

    // ---- 7) concat (bf16) + embed_rxn_map MLP ----
    amk_concat2<<<dim3((BSZ * 2 * DE + 255) / 256), 256, 0, stream>>>(STCAT, M1, CATh, BSZ, DE);
    gemm(CATh, e0, 1024, 128, M2, BSZ, 0, 0);  norm_gelu(M2, M2h, BSZ, 128, en0);
    gemm(M2h, e1, 128, 512, M1, BSZ, 0, 0);    norm_gelu(M1, M1h, BSZ, 512, en1);
    gemm(M1h, e2, 512, 1024, M2, BSZ, 0, 0);   norm_gelu(M2, M2h, BSZ, 1024, en2);
    gemm(M2h, e3, 1024, 3000, OUTh, BSZ, 1, 1);                 // out_rxn (bf16)

    // ---- 8) final: relu(out_rxn @ rxn_gene_map^T + b) -> d_out (16 x 2000 f32) ----
    gemm(OUTh, rgm, NR, NG, d_out, BSZ, 2, 0);
}
